// GCN_13048110645410
// MI455X (gfx1250) — matrix operand order repaired and checked by reading, not compiled, not run on hardware
//
#include <hip/hip_runtime.h>
#include <hip/hip_bf16.h>

#define N_NODES  50000
#define N_EDGES  800000
#define F        64
#define N_CLASSES 10
#define N_GRAPHS 512

typedef __attribute__((ext_vector_type(16))) _Float16 v16h;
typedef __attribute__((ext_vector_type(8)))  _Float16 v8h;
typedef __attribute__((ext_vector_type(8)))  float    v8f;

// ---------------- scatter-add aggregation: agg[dst] += h[src] ----------------
// 16 threads per edge; each thread handles 4 consecutive features (16B gather
// + 4 global_atomic_add_f32). agg (12.8MB) and h (12.8MB) are L2-resident.
__global__ void k_scatter_add(const float* __restrict__ h,
                              const int* __restrict__ src,
                              const int* __restrict__ dst,
                              float* __restrict__ agg)
{
    int tid = blockIdx.x * blockDim.x + threadIdx.x;
    if (tid >= N_EDGES * 16) return;
    int e = tid >> 4;
    int c = (tid & 15) << 2;
    int s = src[e];
    int d = dst[e];
    const float4 v = *(const float4*)(h + (size_t)s * F + c);
    float* p = agg + (size_t)d * F + c;
    atomicAdd(p + 0, v.x);
    atomicAdd(p + 1, v.y);
    atomicAdd(p + 2, v.z);
    atomicAdd(p + 3, v.w);
}

// ---------------- f32 -> f16 conversion of agg and h (8 elems/thread) -------
__global__ void k_cvt(const float* __restrict__ a, const float* __restrict__ h,
                      _Float16* __restrict__ af, _Float16* __restrict__ hf)
{
    int i = (blockIdx.x * blockDim.x + threadIdx.x) * 8;
    if (i >= N_NODES * F) return;
#pragma unroll
    for (int j = 0; j < 8; ++j) {
        af[i + j] = (_Float16)a[i + j];
        hf[i + j] = (_Float16)h[i + j];
    }
}

// ---------------- pack a 64x64 f32 weight into WMMA B-operand layout --------
// out[kt][nt][lane][j] = W[kt*32 + (lane>>4)*8 + (j&7) + (j>>3)*16][nt*16 + (lane&15)]
// (B matrix 32x16 f16 per tile: element j holds K = (lane>>4)*8 + (j&7) + (j>>3)*16 at
//  column N = lane&15 -> per lane 16 contiguous f16 in the packed buffer.)
__global__ void k_pack_w(const float* __restrict__ W, _Float16* __restrict__ out)
{
    int idx = blockIdx.x * blockDim.x + threadIdx.x;
    if (idx >= 2 * 4 * 32 * 16) return;
    int j    = idx & 15;
    int lane = (idx >> 4) & 31;
    int nt   = (idx >> 9) & 3;
    int kt   = idx >> 11;
    int row  = kt * 32 + (lane >> 4) * 8 + (j & 7) + (j >> 3) * 16;
    int col  = nt * 16 + (lane & 15);
    out[idx] = (_Float16)W[row * 64 + col];
}

// ---------------- WMMA GEMM: out = agg@W_rel + h@W_root + b (+ReLU) ---------
// One wave32 per 16-node tile. K=64 -> 2 k-steps of v_wmma_f32_16x16x32_f16;
// N=64 -> 4 n-tiles; two weight matrices -> 16 WMMAs per wave, f32 accum.
__global__ void __launch_bounds__(256)
k_gemm_wmma(const _Float16* __restrict__ Af, const _Float16* __restrict__ Hf,
            const _Float16* __restrict__ Wrel, const _Float16* __restrict__ Wroot,
            const float* __restrict__ bias, float* __restrict__ out, int relu)
{
    int gwave = (blockIdx.x * blockDim.x + threadIdx.x) >> 5;
    int lane  = threadIdx.x & 31;
    if (gwave >= N_NODES / 16) return;          // wave-uniform: EXEC all-1s inside
    int node0 = gwave * 16;
    int laneM = lane & 15;
    int laneH = lane >> 4;

    v8f acc[4] = {};
#pragma unroll
    for (int kt = 0; kt < 2; ++kt) {
        int k0 = kt * 32;
        // A layout (16-bit 16x32): lanes 0-15 = M, hold K = k0+laneH*8..+8 and +16
        const _Float16* arow = Af + (size_t)(node0 + laneM) * F + k0 + laneH * 8;
        const _Float16* hrow = Hf + (size_t)(node0 + laneM) * F + k0 + laneH * 8;
        v8h a_lo = *(const v8h*)(arow);
        v8h a_hi = *(const v8h*)(arow + 16);
        v8h h_lo = *(const v8h*)(hrow);
        v8h h_hi = *(const v8h*)(hrow + 16);
        v16h aA, aH;
#pragma unroll
        for (int i = 0; i < 8; ++i) {
            aA[i] = a_lo[i]; aA[i + 8] = a_hi[i];
            aH[i] = h_lo[i]; aH[i + 8] = h_hi[i];
        }
#pragma unroll
        for (int nt = 0; nt < 4; ++nt) {
            v16h bR = *(const v16h*)(Wrel  + (((kt * 4) + nt) * 32 + lane) * 16);
            acc[nt] = __builtin_amdgcn_wmma_f32_16x16x32_f16(
                false, aA, false, bR, (short)0, acc[nt], false, false);
            v16h bT = *(const v16h*)(Wroot + (((kt * 4) + nt) * 32 + lane) * 16);
            acc[nt] = __builtin_amdgcn_wmma_f32_16x16x32_f16(
                false, aH, false, bT, (short)0, acc[nt], false, false);
        }
    }
    // C/D layout: lanes 0-15 -> M = r, lanes 16-31 -> M = r+8, N = laneM
#pragma unroll
    for (int nt = 0; nt < 4; ++nt) {
        int col  = nt * 16 + laneM;
        float bv = bias[col];
#pragma unroll
        for (int r = 0; r < 8; ++r) {
            int row  = node0 + r + laneH * 8;
            float v  = acc[nt][r] + bv;
            if (relu) v = v > 0.0f ? v : 0.0f;
            out[(size_t)row * F + col] = v;
        }
    }
}

// ---------------- mean pool + final linear ----------------------------------
__global__ void k_count(const int* __restrict__ batch, float* __restrict__ counts)
{
    int i = blockIdx.x * blockDim.x + threadIdx.x;
    if (i >= N_NODES) return;
    atomicAdd(&counts[batch[i]], 1.0f);
}

__global__ void k_pool(const float* __restrict__ h, const int* __restrict__ batch,
                       float* __restrict__ pooled)
{
    int tid = blockIdx.x * blockDim.x + threadIdx.x;
    if (tid >= N_NODES * F) return;
    int i = tid >> 6;
    int f = tid & 63;
    atomicAdd(&pooled[batch[i] * F + f], h[tid]);
}

__global__ void k_final(const float* __restrict__ pooled, const float* __restrict__ counts,
                        const float* __restrict__ Wlin, const float* __restrict__ blin,
                        float* __restrict__ out)
{
    int tid = blockIdx.x * blockDim.x + threadIdx.x;
    if (tid >= N_GRAPHS * N_CLASSES) return;
    int g = tid / N_CLASSES, c = tid % N_CLASSES;
    float s = 0.0f;
#pragma unroll
    for (int k = 0; k < F; ++k) s += pooled[g * F + k] * Wlin[k * N_CLASSES + c];
    float cnt = counts[g];
    cnt = cnt > 1.0f ? cnt : 1.0f;
    out[tid] = s / cnt + blin[c];
}

extern "C" void kernel_launch(void* const* d_in, const int* in_sizes, int n_in,
                              void* d_out, int out_size, void* d_ws, size_t ws_size,
                              hipStream_t stream)
{
    const float* x     = (const float*)d_in[0];
    const int*   edge  = (const int*)d_in[1];
    const int*   batch = (const int*)d_in[2];
    const float* W1r = (const float*)d_in[3];
    const float* b1  = (const float*)d_in[4];
    const float* W1o = (const float*)d_in[5];
    const float* W2r = (const float*)d_in[6];
    const float* b2  = (const float*)d_in[7];
    const float* W2o = (const float*)d_in[8];
    const float* W3r = (const float*)d_in[9];
    const float* b3  = (const float*)d_in[10];
    const float* W3o = (const float*)d_in[11];
    const float* Wl  = (const float*)d_in[12];
    const float* bl  = (const float*)d_in[13];

    const int* src = edge;            // edge_index[0]
    const int* dst = edge + N_EDGES;  // edge_index[1]

    // workspace layout (all offsets 256B-aligned by construction)
    float*     agg  = (float*)d_ws;                          // 12.8 MB
    float*     hA   = agg + (size_t)N_NODES * F;             // 12.8 MB
    float*     hB   = hA  + (size_t)N_NODES * F;             // 12.8 MB
    _Float16*  aggf = (_Float16*)(hB + (size_t)N_NODES * F); //  6.4 MB
    _Float16*  hf   = aggf + (size_t)N_NODES * F;            //  6.4 MB
    _Float16*  pw   = hf   + (size_t)N_NODES * F;            // 6 * 8 KB packed weights
    float*     pooled = (float*)(pw + 6 * 4096);             // 512*64 f32
    float*     counts = pooled + N_GRAPHS * F;               // 512 f32

    // one-time weight repack into WMMA B layout
    k_pack_w<<<16, 256, 0, stream>>>(W1r, pw + 0 * 4096);
    k_pack_w<<<16, 256, 0, stream>>>(W1o, pw + 1 * 4096);
    k_pack_w<<<16, 256, 0, stream>>>(W2r, pw + 2 * 4096);
    k_pack_w<<<16, 256, 0, stream>>>(W2o, pw + 3 * 4096);
    k_pack_w<<<16, 256, 0, stream>>>(W3r, pw + 4 * 4096);
    k_pack_w<<<16, 256, 0, stream>>>(W3o, pw + 5 * 4096);

    const int SC_GRID  = (N_EDGES * 16 + 255) / 256;
    const int CV_GRID  = (N_NODES * F / 8 + 255) / 256;
    const int GM_GRID  = ((N_NODES / 16) * 32 + 255) / 256;

    auto layer = [&](const float* hin, float* hout, const _Float16* wr,
                     const _Float16* wo, const float* b, int relu) {
        hipMemsetAsync(agg, 0, (size_t)N_NODES * F * sizeof(float), stream);
        k_scatter_add<<<SC_GRID, 256, 0, stream>>>(hin, src, dst, agg);
        k_cvt<<<CV_GRID, 256, 0, stream>>>(agg, hin, aggf, hf);
        k_gemm_wmma<<<GM_GRID, 256, 0, stream>>>(aggf, hf, wr, wo, b, hout, relu);
    };

    layer(x,  hA, pw + 0 * 4096, pw + 1 * 4096, b1, 1);
    layer(hA, hB, pw + 2 * 4096, pw + 3 * 4096, b2, 1);
    layer(hB, hA, pw + 4 * 4096, pw + 5 * 4096, b3, 0);

    // global mean pool + final linear
    hipMemsetAsync(pooled, 0, (size_t)(N_GRAPHS * F + N_GRAPHS) * sizeof(float), stream);
    k_count<<<(N_NODES + 255) / 256, 256, 0, stream>>>(batch, counts);
    k_pool<<<(N_NODES * F + 255) / 256, 256, 0, stream>>>(hA, batch, pooled);
    k_final<<<(N_GRAPHS * N_CLASSES + 255) / 256, 256, 0, stream>>>(
        pooled, counts, Wl, bl, (float*)d_out);
}
